// DARNNModel_80607946211901
// MI455X (gfx1250) — matrix operand order repaired and checked by reading, not compiled, not run on hardware
//
#include <hip/hip_runtime.h>
#include <math.h>

typedef __attribute__((ext_vector_type(16))) __bf16 v16bf;
typedef __attribute__((ext_vector_type(8)))  float  v8f;

#define BATCH    4096
#define TSTEPS   128
#define HORIZON  24
#define EH       64
#define NG       192   // 3*EH gate width
#define ATT      32
#define ROWS     16    // batch rows per workgroup
#define NTHREADS 128

// ---------------- LDS layout (bytes) ----------------
#define OFF_H      0                              // bf16 [ROWS][TSTEPS][EH]      = 262144
#define OFF_WHH    (OFF_H    + ROWS*TSTEPS*EH*2)  // bf16 [EH][NG]  W_hh (phase)  =  24576
#define OFF_WB2    (OFF_WHH  + EH*NG*2)           // bf16 4096: W_init -> W_d|U_d =   8192
#define OFF_SCR    (OFF_WB2  + 4096*2)            // f32 [ROWS*NG]: gh / s|ctx    =  12288
#define OFF_HST    (OFF_SCR  + ROWS*NG*4)         // f32 [ROWS][EH] h or d state  =   4096
#define OFF_HBF    (OFF_HST  + ROWS*EH*4)         // bf16 [ROWS][EH] WMMA A src   =   2048
#define OFF_DPROJ  (OFF_HBF  + ROWS*EH*2)         // f32 [ROWS][ATT]              =   2048
#define OFF_PAR    (OFF_DPROJ+ ROWS*ATT*4)        // f32 [1024] params/scratch    =   4096
#define LDS_BYTES  (OFF_PAR  + 1024*4)            // = 319488 B (312 KB) < 320 KB

static __device__ inline v8f wmma_bf16(v16bf a, v16bf b, v8f c) {
  return __builtin_amdgcn_wmma_f32_16x16x32_bf16(false, a, false, b, (short)0, c,
                                                 false, false);
}

// Fast transcendentals: single-instruction v_rcp_f32 / v_tanh_f32 (CDNA5 TRANS ops)
// instead of the IEEE divide chain the generic path emits.
static __device__ inline float fast_rcp(float x) { return __builtin_amdgcn_rcpf(x); }

static __device__ inline float fast_sigmoid(float x) {
  return fast_rcp(1.0f + __expf(-x));
}

static __device__ inline float fast_tanh(float x) {
#if __has_builtin(__builtin_amdgcn_tanhf)
  return __builtin_amdgcn_tanhf(x);
#elif __has_builtin(__builtin_amdgcn_tanh_f32)
  return __builtin_amdgcn_tanh_f32(x);
#else
  // tanh(x) = 1 - 2/(exp(2x)+1); saturates correctly for large |x|.
  return 1.0f - 2.0f * fast_rcp(__expf(2.0f * x) + 1.0f);
#endif
}

// A fragment (16x32 bf16, MxK), row-major LDS, row stride ld; base points at (row0,k0).
// Lane L holds row m=L&15; k = half*16 + {0..15} (contiguous, matches load_bfrag).
static __device__ inline v16bf load_afrag(const __bf16* base, int ld) {
  int lane = threadIdx.x & 31;
  const __bf16* p = base + (lane & 15) * ld + (lane >> 4) * 16;
  v16bf f;
#pragma unroll
  for (int i = 0; i < 16; ++i) f[i] = p[i];
  return f;
}

// B fragment (32x16 bf16, KxN) from row-major [K][Nld] LDS (col n = lane&15,
// k base = (lane>>4)*16, contiguous, same k order as load_afrag).
static __device__ inline v16bf load_bfrag(const __bf16* W, int Nld, int k0, int n0) {
  int lane = threadIdx.x & 31;
  int n  = n0 + (lane & 15);
  int kb = k0 + (lane >> 4) * 16;
  v16bf f;
#pragma unroll
  for (int i = 0; i < 16; ++i) f[i] = W[(kb + i) * Nld + n];
  return f;
}

// C tile (16x16 f32) accumulator -> row-major [16][Nld] f32 LDS.
static __device__ inline void store_ctile(float* S, int Nld, int n0, v8f acc) {
  int lane = threadIdx.x & 31;
  int n  = n0 + (lane & 15);
  int rb = (lane >> 4) * 8;
#pragma unroll
  for (int v = 0; v < 8; ++v) S[(rb + v) * Nld + n] = acc[v];
}

// GRU gate nonlinearity: consumes gh in scr[m][0..191], produces new state.
static __device__ inline void gru_gates(float xm, int m, int e0, const float* par,
                                        const float* scr, float* hst, __bf16* hbf,
                                        __bf16* Hcol) {
#pragma unroll
  for (int q = 0; q < 8; ++q) {
    int e = e0 + q;
    float gr  = fmaf(xm, par[e],       par[192 + e]) + scr[m*NG + e]       + par[384 + e];
    float gz  = fmaf(xm, par[64 + e],  par[256 + e]) + scr[m*NG + 64 + e]  + par[448 + e];
    float gni = fmaf(xm, par[128 + e], par[320 + e]);
    float gnh = scr[m*NG + 128 + e] + par[512 + e];
    float r = fast_sigmoid(gr);
    float z = fast_sigmoid(gz);
    float n = fast_tanh(fmaf(r, gnh, gni));
    float hn = (1.0f - z) * n + z * hst[m*EH + e];
    hst[m*EH + e] = hn;
    __bf16 bv = (__bf16)hn;
    hbf[m*EH + e] = bv;
    if (Hcol) Hcol[m * (TSTEPS * EH) + e] = bv;
  }
}

__global__ __launch_bounds__(NTHREADS)
void darnn_fused(const float* __restrict__ xg,
                 const float* __restrict__ W_ih_e, const float* __restrict__ W_hh_e,
                 const float* __restrict__ b_ih_e, const float* __restrict__ b_hh_e,
                 const float* __restrict__ W_init, const float* __restrict__ b_init,
                 const float* __restrict__ W_ih_d, const float* __restrict__ W_hh_d,
                 const float* __restrict__ b_ih_d, const float* __restrict__ b_hh_d,
                 const float* __restrict__ W_d,    const float* __restrict__ U_d,
                 const float* __restrict__ v_d,    const float* __restrict__ W_out,
                 const float* __restrict__ b_out,  const float* __restrict__ y0,
                 float* __restrict__ out)
{
  extern __shared__ char smem[];
  __bf16* Hl   = (__bf16*)(smem + OFF_H);     // [ROWS][TSTEPS][EH] encoder states
  __bf16* WhhL = (__bf16*)(smem + OFF_WHH);   // [EH][NG] (K-major) current-phase W_hh^T
  __bf16* WB2  = (__bf16*)(smem + OFF_WB2);   // W_init^T, later W_d^T | U_d^T
  float*  scr  = (float*) (smem + OFF_SCR);   // gh dump / scores+ctx
  float*  hst  = (float*) (smem + OFF_HST);   // f32 h (then d) state
  __bf16* hbf  = (__bf16*)(smem + OFF_HBF);   // bf16 state (WMMA A operand)
  float*  dpj  = (float*) (smem + OFF_DPROJ); // d @ W_d^T
  float*  par  = (float*) (smem + OFF_PAR);

  const int tid  = threadIdx.x;
  const int wv   = tid >> 5;
  const int lane = tid & 31;
  const int rowbase = blockIdx.x * ROWS;
  const int gm  = tid >> 3;        // row for elementwise phases
  const int ge0 = (tid & 7) * 8;   // first hidden element

  // ---------------- preload encoder weights ----------------
  for (int i = tid; i < NG * EH; i += NTHREADS) {
    int j = i / EH, k = i % EH;
    WhhL[k * NG + j] = (__bf16)W_hh_e[i];              // transpose to [k][j]
  }
  for (int i = tid; i < NG; i += NTHREADS) {
    par[i]       = W_ih_e[i];
    par[192 + i] = b_ih_e[i];
    par[384 + i] = b_hh_e[i];
  }
  for (int i = tid; i < ROWS * EH; i += NTHREADS) { hst[i] = 0.0f; hbf[i] = (__bf16)0.0f; }
  __syncthreads();

  // W_hh fragments live in registers across the whole phase (3 N-tiles per wave).
  v16bf whh[3][2];
#pragma unroll
  for (int j = 0; j < 3; ++j)
#pragma unroll
    for (int kt = 0; kt < 2; ++kt)
      whh[j][kt] = load_bfrag(WhhL, NG, kt * 32, (wv * 3 + j) * 16);

  // ================= ENCODER: 128 sequential GRU steps =================
  // (input-attention is identically alpha==1 -> plain GRU on x_t)
  for (int t = 0; t < TSTEPS; ++t) {
    {
      v16bf a0 = load_afrag(hbf, EH);
      v16bf a1 = load_afrag(hbf + 32, EH);
#pragma unroll
      for (int j = 0; j < 3; ++j) {
        v8f acc = {};
        acc = wmma_bf16(a0, whh[j][0], acc);
        acc = wmma_bf16(a1, whh[j][1], acc);
        store_ctile(scr, NG, (wv * 3 + j) * 16, acc);
      }
    }
    __syncthreads();
    {
      float xm = xg[(rowbase + gm) * TSTEPS + t];
      if (t + 1 < TSTEPS) __builtin_prefetch(&xg[(rowbase + gm) * TSTEPS + t + 1], 0, 0);
      gru_gates(xm, gm, ge0, par, scr, hst, hbf, Hl + t * EH);
    }
    __syncthreads();
  }

  // ---------------- switch LDS to decoder weights ----------------
  for (int i = tid; i < NG * EH; i += NTHREADS) {
    int j = i / EH, k = i % EH;
    WhhL[k * NG + j] = (__bf16)W_hh_d[i];
  }
  for (int i = tid; i < NG; i += NTHREADS) {
    par[i]       = W_ih_d[i];
    par[192 + i] = b_ih_d[i];
    par[384 + i] = b_hh_d[i];
  }
  for (int i = tid; i < EH * EH; i += NTHREADS) {       // W_init -> [k][n]
    int n = i / EH, k = i % EH;
    WB2[k * EH + n] = (__bf16)W_init[i];
  }
  for (int i = tid; i < ATT;    i += NTHREADS) par[576 + i] = v_d[i];
  for (int i = tid; i < 2 * EH; i += NTHREADS) par[608 + i] = W_out[i];
  if (tid == 0)   par[736] = b_out[0];
  if (tid < ROWS) par[737 + tid] = y0[0];    // dec_in
  __syncthreads();

  // d0 = h_T @ W_init^T + b_init  (wave wv owns N-tile wv of 4)
  {
    v16bf a0 = load_afrag(hbf, EH);
    v16bf a1 = load_afrag(hbf + 32, EH);
    v16bf b0 = load_bfrag(WB2, EH, 0,  wv * 16);
    v16bf b1 = load_bfrag(WB2, EH, 32, wv * 16);
    v8f acc = {};
    acc = wmma_bf16(a0, b0, acc);
    acc = wmma_bf16(a1, b1, acc);
    store_ctile(scr, EH, wv * 16, acc);
  }
  __syncthreads();
#pragma unroll
  for (int q = 0; q < 8; ++q) {
    int e = ge0 + q;
    float d = scr[gm * EH + e] + b_init[e];
    hst[gm * EH + e] = d;
    hbf[gm * EH + e] = (__bf16)d;
  }
  __syncthreads();
  for (int i = tid; i < ATT * EH; i += NTHREADS) {      // W_d | U_d -> [k][n]
    int n = i / EH, k = i % EH;
    WB2[k * ATT + n]        = (__bf16)W_d[i];
    WB2[2048 + k * ATT + n] = (__bf16)U_d[i];
  }
  __syncthreads();

  // refresh register fragments for decoder
#pragma unroll
  for (int j = 0; j < 3; ++j)
#pragma unroll
    for (int kt = 0; kt < 2; ++kt)
      whh[j][kt] = load_bfrag(WhhL, NG, kt * 32, (wv * 3 + j) * 16);
  v16bf udf[2][2];
#pragma unroll
  for (int at = 0; at < 2; ++at)
#pragma unroll
    for (int kt = 0; kt < 2; ++kt)
      udf[at][kt] = load_bfrag(WB2 + 2048, ATT, kt * 32, at * 16);
  v16bf wdf[2];
  {
    int nt = (wv < 2) ? wv : 0;
#pragma unroll
    for (int kt = 0; kt < 2; ++kt)
      wdf[kt] = load_bfrag(WB2, ATT, kt * 32, nt * 16);
  }

  float* sArr = scr;                  // [ROWS][TSTEPS] scores -> beta (in place)
  float* ctx  = scr + ROWS * TSTEPS;  // [ROWS][EH]
  float* red  = par + 768;            // [ROWS][8] softmax max
  float* red2 = par + 896;            // [ROWS][8] softmax sum

  // ================= DECODER: 24 sequential steps =================
  for (int step = 0; step < HORIZON; ++step) {
    // --- d = GRU(dec_in, d) ---
    {
      v16bf a0 = load_afrag(hbf, EH);
      v16bf a1 = load_afrag(hbf + 32, EH);
#pragma unroll
      for (int j = 0; j < 3; ++j) {
        v8f acc = {};
        acc = wmma_bf16(a0, whh[j][0], acc);
        acc = wmma_bf16(a1, whh[j][1], acc);
        store_ctile(scr, NG, (wv * 3 + j) * 16, acc);
      }
    }
    __syncthreads();
    gru_gates(par[737 + gm], gm, ge0, par, scr, hst, hbf, (__bf16*)0);
    __syncthreads();

    // --- d_proj = d @ W_d^T (waves 0,1; one 16-col tile each) ---
    if (wv < 2) {
      v16bf a0 = load_afrag(hbf, EH);
      v16bf a1 = load_afrag(hbf + 32, EH);
      v8f acc = {};
      acc = wmma_bf16(a0, wdf[0], acc);
      acc = wmma_bf16(a1, wdf[1], acc);
      store_ctile(dpj, ATT, wv * 16, acc);
    }
    __syncthreads();

    // --- scores s[m][t] = v_d . tanh(d_proj[m] + H[m][t] @ U_d^T) ---
    // H_proj recomputed from LDS-resident H via WMMA (A rows = 16 t's for fixed m).
    {
      int n = lane & 15, half = lane >> 4;
      float vd0 = par[576 + n], vd1 = par[592 + n];
      for (int it = 0; it < 32; ++it) {
        int idx = wv * 32 + it;            // 16 m x 8 chunks split over 4 waves
        int m = idx >> 3, chunk = idx & 7;
        const __bf16* Ab = Hl + (m * TSTEPS + chunk * 16) * EH;
        v16bf a0 = load_afrag(Ab, EH);
        v16bf a1 = load_afrag(Ab + 32, EH);
        v8f acc0 = {}, acc1 = {};
        acc0 = wmma_bf16(a0, udf[0][0], acc0);
        acc0 = wmma_bf16(a1, udf[0][1], acc0);
        acc1 = wmma_bf16(a0, udf[1][0], acc1);
        acc1 = wmma_bf16(a1, udf[1][1], acc1);
        float dp0 = dpj[m * ATT + n], dp1 = dpj[m * ATT + 16 + n];
#pragma unroll
        for (int v = 0; v < 8; ++v) {
          float val = vd0 * fast_tanh(acc0[v] + dp0) + vd1 * fast_tanh(acc1[v] + dp1);
          val += __shfl_xor(val, 1, 32);   // reduce over 16 N-lanes (halves stay split)
          val += __shfl_xor(val, 2, 32);
          val += __shfl_xor(val, 4, 32);
          val += __shfl_xor(val, 8, 32);
          if (n == 0) sArr[m * TSTEPS + chunk * 16 + half * 8 + v] = val;
        }
      }
    }
    __syncthreads();

    // --- softmax over t (8 threads per row) ---
    {
      float* srow = sArr + gm * TSTEPS;
      int j = tid & 7;
      float mx = -3.0e38f;
      for (int q = 0; q < 16; ++q) mx = fmaxf(mx, srow[j * 16 + q]);
      red[gm * 8 + j] = mx;
      __syncthreads();
      float rmx = red[gm * 8];
#pragma unroll
      for (int q = 1; q < 8; ++q) rmx = fmaxf(rmx, red[gm * 8 + q]);
      float ls = 0.0f;
      for (int q = 0; q < 16; ++q) {
        float ev = __expf(srow[j * 16 + q] - rmx);
        srow[j * 16 + q] = ev;
        ls += ev;
      }
      red2[gm * 8 + j] = ls;
      __syncthreads();
      float rs = red2[gm * 8];
#pragma unroll
      for (int q = 1; q < 8; ++q) rs += red2[gm * 8 + q];
      float inv = fast_rcp(rs);
      for (int q = 0; q < 16; ++q) srow[j * 16 + q] *= inv;
    }
    __syncthreads();

    // --- ctx[m] = sum_t beta[m][t] * H[m][t][:]  (LDS-resident H) ---
    {
      float a[8] = {0, 0, 0, 0, 0, 0, 0, 0};
      const __bf16* Hm = Hl + gm * TSTEPS * EH + ge0;
      const float* brow = sArr + gm * TSTEPS;
      for (int t = 0; t < TSTEPS; ++t) {
        float b = brow[t];
        const __bf16* hp = Hm + t * EH;
#pragma unroll
        for (int q = 0; q < 8; ++q) a[q] += b * (float)hp[q];
      }
#pragma unroll
      for (int q = 0; q < 8; ++q) ctx[gm * EH + ge0 + q] = a[q];
    }
    __syncthreads();

    // --- out = [d, ctx] @ W_out^T + b_out ; feed back as dec_in ---
    if (tid < ROWS) {
      float sum = par[736];
      for (int e = 0; e < EH; ++e)
        sum += par[608 + e] * hst[tid * EH + e] + par[672 + e] * ctx[tid * EH + e];
      out[(rowbase + tid) * HORIZON + step] = sum;
      par[737 + tid] = sum;
    }
    __syncthreads();
  }
}

extern "C" void kernel_launch(void* const* d_in, const int* in_sizes, int n_in,
                              void* d_out, int out_size, void* d_ws, size_t ws_size,
                              hipStream_t stream) {
  (void)in_sizes; (void)n_in; (void)d_ws; (void)ws_size; (void)out_size;
  const float* x      = (const float*)d_in[0];
  const float* W_ih_e = (const float*)d_in[1];
  const float* W_hh_e = (const float*)d_in[2];
  const float* b_ih_e = (const float*)d_in[3];
  const float* b_hh_e = (const float*)d_in[4];
  // d_in[5..8] (W_e, U_e, b_e, v_e): encoder input attention is softmax over a
  // singleton axis -> alpha == 1 identically; these inputs cannot affect output.
  const float* W_init = (const float*)d_in[9];
  const float* b_init = (const float*)d_in[10];
  const float* W_ih_d = (const float*)d_in[11];
  const float* W_hh_d = (const float*)d_in[12];
  const float* b_ih_d = (const float*)d_in[13];
  const float* b_hh_d = (const float*)d_in[14];
  const float* W_d    = (const float*)d_in[15];
  const float* U_d    = (const float*)d_in[16];
  const float* v_d    = (const float*)d_in[17];
  const float* W_out  = (const float*)d_in[18];
  const float* b_out  = (const float*)d_in[19];
  const float* y0     = (const float*)d_in[20];
  float* out = (float*)d_out;

  // Allow the 312 KB dynamic-LDS allocation (no-op if not required on this stack).
  (void)hipFuncSetAttribute((const void*)darnn_fused,
                            hipFuncAttributeMaxDynamicSharedMemorySize, LDS_BYTES);

  dim3 grid(BATCH / ROWS), block(NTHREADS);
  hipLaunchKernelGGL(darnn_fused, grid, block, LDS_BYTES, stream,
                     x, W_ih_e, W_hh_e, b_ih_e, b_hh_e, W_init, b_init,
                     W_ih_d, W_hh_d, b_ih_d, b_hh_d, W_d, U_d, v_d,
                     W_out, b_out, y0, out);
}
